// PointNetSetAbstractionMsg_70772471103961
// MI455X (gfx1250) — matrix operand order repaired and checked by reading, not compiled, not run on hardware
//
#include <hip/hip_runtime.h>
#include <hip/hip_bf16.h>

// ---------------- problem constants (match reference) ----------------
#define B_  8
#define N_  8192
#define S_  2048
#define EPSC 1e-5f

typedef __attribute__((ext_vector_type(16))) _Float16 v16h;
typedef __attribute__((ext_vector_type(8)))  _Float16 v8h;
typedef __attribute__((ext_vector_type(8)))  float    v8f;

// ==================================================================
// FPS: one workgroup (1024 threads) per batch. dist[] lives in LDS.
// ==================================================================
__global__ __launch_bounds__(1024)
void fps_kernel(const float* __restrict__ xyz, int* __restrict__ fpsIdx) {
  __shared__ float sdist[N_];
  __shared__ float rv[32];
  __shared__ int   ri[32];
  __shared__ int   sFar;
  const int b   = blockIdx.x;
  const int tid = threadIdx.x;
  const float* xb = xyz + (size_t)b * 3 * N_;
  for (int j = tid; j < N_; j += 1024) sdist[j] = 1e10f;
  __syncthreads();
  int far = 0;
  for (int it = 0; it < S_; ++it) {
    if (tid == 0) fpsIdx[b * S_ + it] = far;
    const float cx = xb[far], cy = xb[N_ + far], cz = xb[2 * N_ + far];
    float bestv = -1.f; int besti = 0;
    for (int j = tid; j < N_; j += 1024) {
      float dx = xb[j] - cx, dy = xb[N_ + j] - cy, dz = xb[2 * N_ + j] - cz;
      float d  = dx * dx + dy * dy + dz * dz;
      float dd = fminf(sdist[j], d);
      sdist[j] = dd;
      if (dd > bestv) { bestv = dd; besti = j; }
    }
    // wave32 argmax reduce (prefer larger value, then smaller index)
    for (int off = 16; off > 0; off >>= 1) {
      float ov = __shfl_down(bestv, off);
      int   oi = __shfl_down(besti, off);
      if (ov > bestv || (ov == bestv && oi < besti)) { bestv = ov; besti = oi; }
    }
    if ((tid & 31) == 0) { rv[tid >> 5] = bestv; ri[tid >> 5] = besti; }
    __syncthreads();
    if (tid == 0) {
      float bv = rv[0]; int bi = ri[0];
      for (int k = 1; k < 32; ++k)
        if (rv[k] > bv || (rv[k] == bv && ri[k] < bi)) { bv = rv[k]; bi = ri[k]; }
      sFar = bi;
    }
    __syncthreads();
    far = sFar;
  }
}

// ==================================================================
// Gather sampled centers; write ws copy [B*S][3] and d_out (B,3,S).
// ==================================================================
__global__ void gather_newxyz_kernel(const float* __restrict__ xyz,
                                     const int* __restrict__ fpsIdx,
                                     float* __restrict__ nxyz,
                                     float* __restrict__ outXyz) {
  int i = blockIdx.x * blockDim.x + threadIdx.x;
  if (i >= B_ * S_) return;
  int b = i / S_, s = i % S_;
  int j = fpsIdx[i];
  for (int c = 0; c < 3; ++c) {
    float v = xyz[(size_t)b * 3 * N_ + c * N_ + j];
    nxyz[(size_t)i * 3 + c] = v;
    outXyz[(size_t)b * 3 * S_ + c * S_ + s] = v;
  }
}

// ==================================================================
// Ball query: one wave per center; ordered first-K via ballot/popc.
// Matches reference (keep sqr<=r^2 in index order, pad with first hit).
// ==================================================================
template <int K>
__global__ __launch_bounds__(256)
void ball_kernel(const float* __restrict__ xyz, const float* __restrict__ nxyz,
                 int* __restrict__ out, float r2) {
  int wid  = blockIdx.x * 8 + (threadIdx.x >> 5);
  int lane = threadIdx.x & 31;
  int b = wid / S_, s = wid % S_;
  const float* xb = xyz + (size_t)b * 3 * N_;
  float cx = nxyz[(size_t)wid * 3 + 0];
  float cy = nxyz[(size_t)wid * 3 + 1];
  float cz = nxyz[(size_t)wid * 3 + 2];
  int* o = out + (size_t)wid * K;
  int cnt = 0, firstIdx = -1;
  for (int j0 = 0; j0 < N_; j0 += 32) {
    int j = j0 + lane;
    float dx = xb[j] - cx, dy = xb[N_ + j] - cy, dz = xb[2 * N_ + j] - cz;
    float d  = dx * dx + dy * dy + dz * dz;
    unsigned m = (unsigned)__ballot(d <= r2);
    if (m) {
      if (firstIdx < 0) firstIdx = j0 + __builtin_ctz(m);
      int pos = cnt + __popc(m & ((1u << lane) - 1u));
      if ((d <= r2) && pos < K) o[pos] = j;
      cnt += __popc(m);
      if (cnt >= K) break;
    }
  }
  if (firstIdx < 0) firstIdx = 0;
  for (int p = cnt + lane; p < K; p += 32) o[p] = firstIdx;
}

// ==================================================================
// Weight prep: fold BN into W/bias, emit f16 WMMA B-fragment blob.
// B 32x16 f16 layout: lane L holds column n=L%16; element e -> k = (e&7) + (e>=8?16:0) + (L<16?0:8).
// ==================================================================
__global__ void prep_layer_kernel(const float* __restrict__ W, const float* __restrict__ bias,
                                  const float* __restrict__ gamma, const float* __restrict__ beta,
                                  const float* __restrict__ mean, const float* __restrict__ var,
                                  int cinA, int kc, int ct,
                                  _Float16* __restrict__ wfrag, float* __restrict__ bfold) {
  int i = blockIdx.x * blockDim.x + threadIdx.x;
  int cout = ct * 16;
  if (i < cout) {
    float sc = gamma[i] * rsqrtf(var[i] + EPSC);
    bfold[i] = (bias[i] - mean[i]) * sc + beta[i];
  }
  int total = kc * ct * 512;
  if (i >= total) return;
  int e    = i & 15;
  int lane = (i >> 4) & 31;
  int qt   = i >> 9;
  int t    = qt % ct;
  int q    = qt / ct;
  int n    = t * 16 + (lane & 15);
  int k    = q * 32 + (e & 7) + ((e >> 3) << 4) + ((lane < 16) ? 0 : 8);
  float v = 0.f;
  if (k < cinA) {
    float sc = gamma[n] * rsqrtf(var[n] + EPSC);
    v = W[n * cinA + k] * sc;
  }
  wfrag[i] = (_Float16)v;
}

// ==================================================================
// Fused grouped-MLP + maxpool, WMMA f32_16x16x32_f16.
// Each WG: 128 rows (TS centers x K neighbors), 8 waves, 1 row-tile/wave.
// ==================================================================
template <int SC> struct Cfg;
template <> struct Cfg<0> {
  static constexpr int K=16, TS=8, WTOT=4096,  BTOT=128, WGOFF=0,     BGOFF=0;
  static constexpr int KC0=1, CT0=2, KC1=1, CT1=2, KC2=1, CT2=4;
  static constexpr int WO0=0, WO1=1024, WO2=2048;
  static constexpr int BO0=0, BO1=32,   BO2=64;
  static constexpr int CLAST=64, CHOFF=0;
};
template <> struct Cfg<1> {
  static constexpr int K=32, TS=4, WTOT=14336, BTOT=256, WGOFF=4096,  BGOFF=128;
  static constexpr int KC0=1, CT0=4, KC1=2, CT1=4, KC2=2, CT2=8;
  static constexpr int WO0=0, WO1=2048, WO2=6144;
  static constexpr int BO0=0, BO1=64,   BO2=128;
  static constexpr int CLAST=128, CHOFF=64;
};
template <> struct Cfg<2> {
  static constexpr int K=64, TS=2, WTOT=20480, BTOT=288, WGOFF=18432, BGOFF=384;
  static constexpr int KC0=1, CT0=4, KC1=2, CT1=6, KC2=3, CT2=8;
  static constexpr int WO0=0, WO1=2048, WO2=8192;
  static constexpr int BO0=0, BO1=64,   BO2=160;
  static constexpr int CLAST=128, CHOFF=192;
};

template <int KC, int CT>
__device__ __forceinline__ void gemm_layer(const _Float16* __restrict__ src,
                                           _Float16* __restrict__ dst,
                                           const _Float16* __restrict__ wf,
                                           const float* __restrict__ bias, int tid) {
  const int wave = tid >> 5;
  const int lane = tid & 31;
  const int hsel = (lane < 16) ? 0 : 1;     // lane half select
  v8f acc[CT] = {};
  const _Float16* arow = src + (wave * 16 + (lane & 15)) * 128;
  const v16h* wfv = (const v16h*)wf;
  #pragma unroll
  for (int q = 0; q < KC; ++q) {
    // A 16x32 f16 fragment (ISA 7.12.2): lanes<16 hold K {0..7,16..23}, lanes>=16 hold {8..15,24..31}
    v8h alo = *(const v8h*)(arow + q * 32 + hsel * 8);
    v8h ahi = *(const v8h*)(arow + q * 32 + 16 + hsel * 8);
    v16h a = __builtin_shufflevector(alo, ahi, 0,1,2,3,4,5,6,7,8,9,10,11,12,13,14,15);
    #pragma unroll
    for (int t = 0; t < CT; ++t) {
      v16h bf = wfv[(q * CT + t) * 32 + lane];
      acc[t] = __builtin_amdgcn_wmma_f32_16x16x32_f16(
          /*neg_a=*/false, a, /*neg_b=*/false, bf,
          /*c_mod=*/(short)0, acc[t], /*reuse_a=*/false, /*reuse_b=*/false);
    }
  }
  // epilogue: bias + ReLU, store f16. D VGPR r: row = wave*16 + r + (lane<16?0:8)
  const int rbase = wave * 16 + hsel * 8;
  const int ncol  = lane & 15;
  #pragma unroll
  for (int t = 0; t < CT; ++t) {
    int n = t * 16 + ncol;
    float bb = bias[n];
    v8f a = acc[t];
    #pragma unroll
    for (int r = 0; r < 8; ++r) {
      float v = fmaxf(a[r] + bb, 0.f);
      dst[(rbase + r) * 128 + n] = (_Float16)v;
    }
  }
}

template <int SC>
__global__ __launch_bounds__(256)
void msg_mlp_kernel(const float* __restrict__ xyz, const float* __restrict__ points,
                    const float* __restrict__ nxyz, const int* __restrict__ ballidx,
                    const _Float16* __restrict__ wfragG, const float* __restrict__ biasG,
                    float* __restrict__ outPts) {
  using C = Cfg<SC>;
  constexpr int K = C::K, TS = C::TS;
  __shared__ __align__(32) _Float16 sW[C::WTOT];
  __shared__ float sBias[C::BTOT];
  __shared__ float sCtr[TS * 3];
  __shared__ __align__(16) _Float16 buf0[128 * 128];
  __shared__ __align__(16) _Float16 buf1[128 * 128];

  const int tid = threadIdx.x;
  const int wg  = blockIdx.x;
  const int b   = wg / (S_ / TS);
  const int s0  = (wg % (S_ / TS)) * TS;

  { // stage weights / bias / centers into LDS
    const uint4* s4 = (const uint4*)(wfragG + C::WGOFF);
    uint4* d4 = (uint4*)sW;
    for (int i = tid; i < C::WTOT / 8; i += 256) d4[i] = s4[i];
    for (int i = tid; i < C::BTOT; i += 256) sBias[i] = biasG[C::BGOFF + i];
    for (int i = tid; i < TS * 3; i += 256) sCtr[i] = nxyz[((size_t)b * S_ + s0) * 3 + i];
  }
  const int* bidx = ballidx + ((size_t)b * S_ + s0) * K;
  __syncthreads();

  // build grouped input X: 128 rows x 32 cols (ch 0..5 feats, 6..8 rel xyz, 9..31 zero)
  for (int i = tid; i < 128 * 32; i += 256) {
    int m = i >> 5, c = i & 31;
    int ts = m / K, kk = m % K;
    int pj = bidx[ts * K + kk];
    float v = 0.f;
    if (c < 6)       v = points[(size_t)b * 6 * N_ + c * N_ + pj];
    else if (c < 9)  v = xyz[(size_t)b * 3 * N_ + (c - 6) * N_ + pj] - sCtr[ts * 3 + (c - 6)];
    buf0[m * 128 + c] = (_Float16)v;
  }
  __syncthreads();

  gemm_layer<C::KC0, C::CT0>(buf0, buf1, sW + C::WO0, sBias + C::BO0, tid);
  __syncthreads();
  gemm_layer<C::KC1, C::CT1>(buf1, buf0, sW + C::WO1, sBias + C::BO1, tid);
  __syncthreads();
  gemm_layer<C::KC2, C::CT2>(buf0, buf1, sW + C::WO2, sBias + C::BO2, tid);
  __syncthreads();

  // max-pool over K neighbors, write (B, 320, S) slice
  for (int p = tid; p < TS * C::CLAST; p += 256) {
    int ts = p / C::CLAST, ch = p % C::CLAST;
    float mx = -1e30f;
    for (int kk = 0; kk < K; ++kk)
      mx = fmaxf(mx, (float)buf1[(ts * K + kk) * 128 + ch]);
    outPts[(size_t)b * 320 * S_ + (size_t)(C::CHOFF + ch) * S_ + (s0 + ts)] = mx;
  }
}

// ==================================================================
// Host launcher
// ==================================================================
extern "C" void kernel_launch(void* const* d_in, const int* in_sizes, int n_in,
                              void* d_out, int out_size, void* d_ws, size_t ws_size,
                              hipStream_t stream) {
  const float* xyz    = (const float*)d_in[0];
  const float* points = (const float*)d_in[1];

  // workspace carve (bytes)
  char* w = (char*)d_ws;
  int*      fpsIdx = (int*)(w + 0);              //  64 KB
  float*    nxyz   = (float*)(w + 65536);        // 192 KB
  int*      ball0  = (int*)(w + 262144);         //   1 MB
  int*      ball1  = (int*)(w + 1310720);        //   2 MB
  int*      ball2  = (int*)(w + 3407872);        //   4 MB
  _Float16* wfrag  = (_Float16*)(w + 7602176);   //  76 KB
  float*    bfold  = (float*)(w + 7680000);      // 2.7 KB

  float* outXyz = (float*)d_out;                 // (B,3,S)
  float* outPts = outXyz + (size_t)B_ * 3 * S_;  // (B,320,S)

  // 1) FPS + center gather
  fps_kernel<<<B_, 1024, 0, stream>>>(xyz, fpsIdx);
  gather_newxyz_kernel<<<(B_ * S_ + 255) / 256, 256, 0, stream>>>(xyz, fpsIdx, nxyz, outXyz);

  // 2) weight folding + fragment packing (9 layers)
  static const int KC[9]   = {1,1,1, 1,2,2, 1,2,3};
  static const int CT[9]   = {2,2,4, 4,4,8, 4,6,8};
  static const int CINA[9] = {9,32,32, 9,64,64, 9,64,96};
  static const int WOFF[9] = {0,1024,2048, 4096,6144,10240, 18432,20480,26624};
  static const int BOFF[9] = {0,32,64, 128,192,256, 384,448,544};
  for (int ly = 0; ly < 9; ++ly) {
    const float* W  = (const float*)d_in[2 + ly * 6 + 0];
    const float* bb = (const float*)d_in[2 + ly * 6 + 1];
    const float* gg = (const float*)d_in[2 + ly * 6 + 2];
    const float* be = (const float*)d_in[2 + ly * 6 + 3];
    const float* mu = (const float*)d_in[2 + ly * 6 + 4];
    const float* vr = (const float*)d_in[2 + ly * 6 + 5];
    int total = KC[ly] * CT[ly] * 512;
    prep_layer_kernel<<<(total + 255) / 256, 256, 0, stream>>>(
        W, bb, gg, be, mu, vr, CINA[ly], KC[ly], CT[ly],
        wfrag + WOFF[ly], bfold + BOFF[ly]);
  }

  // 3) ball query (one wave per center)
  const int ballBlocks = (B_ * S_) / 8;
  ball_kernel<16><<<ballBlocks, 256, 0, stream>>>(xyz, nxyz, ball0, 0.1f * 0.1f);
  ball_kernel<32><<<ballBlocks, 256, 0, stream>>>(xyz, nxyz, ball1, 0.2f * 0.2f);
  ball_kernel<64><<<ballBlocks, 256, 0, stream>>>(xyz, nxyz, ball2, 0.4f * 0.4f);

  // 4) fused gather + MLP (WMMA) + maxpool per scale
  msg_mlp_kernel<0><<<(B_ * S_) / Cfg<0>::TS, 256, 0, stream>>>(xyz, points, nxyz, ball0, wfrag, bfold, outPts);
  msg_mlp_kernel<1><<<(B_ * S_) / Cfg<1>::TS, 256, 0, stream>>>(xyz, points, nxyz, ball1, wfrag, bfold, outPts);
  msg_mlp_kernel<2><<<(B_ * S_) / Cfg<2>::TS, 256, 0, stream>>>(xyz, points, nxyz, ball2, wfrag, bfold, outPts);
}
